// GaussianUpsampling_1159641170496
// MI455X (gfx1250) — matrix operand order repaired and checked by reading, not compiled, not run on hardware
//
#include <hip/hip_runtime.h>

typedef _Float16 f16;
typedef __attribute__((ext_vector_type(16))) _Float16 v16h;
typedef __attribute__((ext_vector_type(8)))  _Float16 v8h;
typedef __attribute__((ext_vector_type(8)))  float    v8f;

#define NB      16
#define T_TEXT  512
#define ADIM    256
#define T_FEATS 4096
#define DELTA   0.1f
#define BSTRIDE 520   // padded halves per LDS row (multiple of 8 -> 16B aligned rows)

// ---------------- kernel 1: token centers c = cumsum(ds) - ds/2, mask-encoded ----
__global__ void centers_kernel(const float* __restrict__ ds,
                               const unsigned char* __restrict__ d_masks,
                               float* __restrict__ cw) {
  __shared__ float s[T_TEXT];
  const int b = blockIdx.x;
  const int t = threadIdx.x;
  const float d = ds[b * T_TEXT + t];
  s[t] = d;
  __syncthreads();
  for (int off = 1; off < T_TEXT; off <<= 1) {
    float v = (t >= off) ? s[t - off] : 0.0f;
    __syncthreads();
    s[t] += v;
    __syncthreads();
  }
  float c = s[t] - 0.5f * d;
  if (!d_masks[b * T_TEXT + t]) c = 1e30f;  // (t-c)^2 -> inf -> energy -inf -> p = 0
  cw[b * T_TEXT + t] = c;
}

// ---------------- kernel 2: fused softmax + WMMA GEMM --------------------------
__global__ void __launch_bounds__(256)
gauss_wmma_kernel(const float* __restrict__ hs,
                  const unsigned char* __restrict__ h_masks,
                  const float* __restrict__ cw,
                  float* __restrict__ out) {
  extern __shared__ char smem[];
  f16*   hsT = (f16*)smem;                                     // [ADIM][BSTRIDE] halves
  float* c_s = (float*)(smem + ADIM * BSTRIDE * sizeof(f16));  // [T_TEXT]

  const int blk  = blockIdx.x;
  const int b    = blk >> 5;     // batch
  const int fblk = blk & 31;     // block of 128 frames
  const int tid  = threadIdx.x;
  const int lane = tid & 31;
  const int wave = tid >> 5;

  // stage token centers
  c_s[tid]       = cw[b * T_TEXT + tid];
  c_s[tid + 256] = cw[b * T_TEXT + tid + 256];

  // stage hs[b] -> f16, transposed [n][k], padded stride
  const float4* hs4 = (const float4*)(hs + (size_t)b * T_TEXT * ADIM);
  for (int idx = tid; idx < T_TEXT * (ADIM / 4); idx += 256) {
    const int k = idx >> 6;          // token row
    const int n = (idx & 63) << 2;   // 4 consecutive adim columns
    float4 v = hs4[idx];
    hsT[(n + 0) * BSTRIDE + k] = (f16)v.x;
    hsT[(n + 1) * BSTRIDE + k] = (f16)v.y;
    hsT[(n + 2) * BSTRIDE + k] = (f16)v.z;
    hsT[(n + 3) * BSTRIDE + k] = (f16)v.w;
  }
  __syncthreads();

  // per-wave: 16-frame output tile
  const int m     = lane & 15;                 // row within tile / column within N-tile
  const int f0    = fblk * 128 + wave * 16;    // first frame of this tile
  const int f     = f0 + m;
  const float t   = h_masks[(size_t)b * T_FEATS + f] ? (float)f : 0.0f;
  const int kbase = (lane & 16) >> 1;          // A lane-half K offset: 0 or 8

  // pass 1: row max over this lane's K subset (k where (k&8)==kbase)
  float mx = -__builtin_inff();
  for (int kk = 0; kk < 256; ++kk) {
    const int k = ((kk & ~7) << 1) + kbase + (kk & 7);
    const float d = t - c_s[k];
    mx = fmaxf(mx, -DELTA * d * d);
  }
  mx = fmaxf(mx, __shfl_xor(mx, 16, 32));      // pair lane m with lane m+16

  // pass 2: row sum of exp(e - max)
  float sum = 0.0f;
  for (int kk = 0; kk < 256; ++kk) {
    const int k = ((kk & ~7) << 1) + kbase + (kk & 7);
    const float d = t - c_s[k];
    sum += __expf(-DELTA * d * d - mx);
  }
  sum += __shfl_xor(sum, 16, 32);
  const float inv = 1.0f / sum;

  // accumulators: 16 N-tiles of 16x16 f32
  v8f acc[16];
  {
    v8f z = {};
#pragma unroll
    for (int nt = 0; nt < 16; ++nt) acc[nt] = z;
  }

  const int bkoff = (lane & 16) >> 1;          // B lane-half K offset: 0 or 8

#pragma unroll 1
  for (int ks = 0; ks < 16; ++ks) {
    // build A fragment (16x32 f16): half i -> K = ks*32 + (i>>3)*16 + kbase + (i&7)
    v16h a;
#pragma unroll
    for (int i = 0; i < 16; ++i) {
      const int k = ks * 32 + ((i >> 3) << 4) + kbase + (i & 7);
      const float d = t - c_s[k];
      a[i] = (f16)(__expf(-DELTA * d * d - mx) * inv);
    }
    const int koff = ks * 32 + bkoff;
#pragma unroll
    for (int nt = 0; nt < 16; ++nt) {
      // B fragment (32x16 f16): lane column n = nt*16 + m, halves K = koff..koff+7, koff+16..koff+23
      const f16* bp = &hsT[(nt * 16 + m) * BSTRIDE + koff];
      v8h lo = *(const v8h*)bp;
      v8h hi = *(const v8h*)(bp + 16);
      v16h bf = __builtin_shufflevector(lo, hi, 0, 1, 2, 3, 4, 5, 6, 7,
                                                8, 9, 10, 11, 12, 13, 14, 15);
      acc[nt] = __builtin_amdgcn_wmma_f32_16x16x32_f16(
          /*neg_a=*/false, a, /*neg_b=*/false, bf,
          /*c_mod=*/(short)0, acc[nt], /*reuse_a=*/false, /*reuse_b=*/false);
    }
  }

  // writeback: C/D layout — VGPR r: M = r (lanes 0-15) or r+8 (lanes 16-31), N = lane&15
  float* op = out + ((size_t)b * T_FEATS + f0) * ADIM;
  const int mb = (lane & 16) >> 1;  // 0 or 8
#pragma unroll
  for (int nt = 0; nt < 16; ++nt) {
#pragma unroll
    for (int r = 0; r < 8; ++r) {
      op[(size_t)(mb + r) * ADIM + nt * 16 + m] = acc[nt][r];
    }
  }
}

extern "C" void kernel_launch(void* const* d_in, const int* in_sizes, int n_in,
                              void* d_out, int out_size, void* d_ws, size_t ws_size,
                              hipStream_t stream) {
  const float*         hs      = (const float*)d_in[0];
  const float*         ds      = (const float*)d_in[1];
  const unsigned char* h_masks = (const unsigned char*)d_in[2];
  const unsigned char* d_masks = (const unsigned char*)d_in[3];
  float* out = (float*)d_out;
  float* cw  = (float*)d_ws;  // 16*512 floats of scratch for token centers

  centers_kernel<<<NB, T_TEXT, 0, stream>>>(ds, d_masks, cw);

  const size_t lds_bytes = (size_t)ADIM * BSTRIDE * sizeof(f16) + T_TEXT * sizeof(float);
  gauss_wmma_kernel<<<NB * (T_FEATS / 128), 256, lds_bytes, stream>>>(hs, h_masks, cw, out);
}
